// GraphCounteRGAN_82471962018372
// MI455X (gfx1250) — compile-verified, matrix operand order checked by reading, not run on hardware
//
#include <hip/hip_runtime.h>
#include <hip/hip_bf16.h>

// ---------------------------------------------------------------------------
// GraphCounteRGAN forward on MI455X (gfx1250, wave32).
// GEMMs via V_WMMA_F32_16X16X32_F16: one wave computes a 16x64 output strip
// (4 accumulators share each A fragment); K fully unrolled. Weights are
// pre-packed into WMMA B-fragment layout (f16) so each B fragment is one
// 32-byte load. Edge propagation via L2-resident atomic scatter-add.
// ---------------------------------------------------------------------------

typedef __attribute__((ext_vector_type(16))) _Float16 v16h;
typedef __attribute__((ext_vector_type(8)))  float    v8f;

#ifndef LANES
#define LANES 32
#endif

// ---------------- utility fills ----------------
__global__ void fill_kernel(float* __restrict__ p, long long n, float v) {
  long long i = (long long)blockIdx.x * blockDim.x + threadIdx.x;
  long long stride = (long long)gridDim.x * blockDim.x;
  for (; i < n; i += stride) p[i] = v;
}

// ---------------- degree scatter: deg[col[e]] += w[e] ----------------
__global__ void deg_kernel(const int* __restrict__ col, const float* __restrict__ w,
                           float* __restrict__ deg, int E) {
  long long i = (long long)blockIdx.x * blockDim.x + threadIdx.x;
  long long stride = (long long)gridDim.x * blockDim.x;
  for (; i < E; i += stride) atomicAdd(&deg[col[i]], w[i]);
}

// ---------------- dinv[i] = deg>0 ? rsqrt(deg) : 0 (in place) ----------------
__global__ void dinv_kernel(float* __restrict__ deg, int n) {
  long long i = (long long)blockIdx.x * blockDim.x + threadIdx.x;
  long long stride = (long long)gridDim.x * blockDim.x;
  for (; i < n; i += stride) {
    float d = deg[i];
    deg[i] = (d > 0.0f) ? rsqrtf(d) : 0.0f;
  }
}

// ---------------- norm[e] = dinv[row]*w*dinv[col] ----------------
__global__ void norm_kernel(const int* __restrict__ row, const int* __restrict__ col,
                            const float* __restrict__ w, const float* __restrict__ dinv,
                            float* __restrict__ norm, int E) {
  long long i = (long long)blockIdx.x * blockDim.x + threadIdx.x;
  long long stride = (long long)gridDim.x * blockDim.x;
  for (; i < E; i += stride)
    norm[i] = dinv[row[i]] * w[i] * dinv[col[i]];
}

// ---------------- pack B[K,N] f32 -> WMMA B-fragment layout f16 ----------------
// Fragment (s = k-step, t = 16-col subtile): 32 lanes x 16 f16.
// Lane l (n = l&15, half = l>>4) holds B[s*32 + half*16 + j][t*16 + n], j=0..15.
// Stored at Bp[((s*(N/16) + t)*32 + l)*16 + j].
__global__ void pack_b_kernel(const float* __restrict__ B, _Float16* __restrict__ Bp,
                              int N, int K) {
  const int fragsN = N >> 4;
  const int total = (K >> 5) * fragsN * 32;      // one thread per (frag, lane)
  const int tid = blockIdx.x * blockDim.x + threadIdx.x;
  if (tid >= total) return;
  const int lane = tid & 31;
  const int frag = tid >> 5;
  const int t = frag % fragsN;
  const int s = frag / fragsN;
  const int kbase = s * 32 + (lane >> 4) * 16;
  const int colB = t * 16 + (lane & 15);
  _Float16* dst = Bp + (size_t)tid * 16;
#pragma unroll
  for (int j = 0; j < 16; ++j)
    dst[j] = (_Float16)B[(size_t)(kbase + j) * N + colB];
}

// ---------------- WMMA GEMM: C[M,N] = (A (+A2)) [M,K] x B[K,N] ----------------
// One wave computes a 16x64 strip of C: 4 accumulators share each A fragment.
// K is a template constant (fully unrolled); HAS_A2 folds enc = z + features.
// Bp is the pre-packed f16 fragment array. M % 16 == 0, N % 64 == 0.
template <int K, bool HAS_A2>
__global__ void gemm_wmma_kernel(const float* __restrict__ A, const float* __restrict__ A2,
                                 const v16h* __restrict__ Bp, float* __restrict__ C,
                                 int M, int N) {
  const int wave = threadIdx.x / LANES;
  const int lane = threadIdx.x & (LANES - 1);
  const int stripsN = N >> 6;                       // 64-wide output strips
  const int fragsN  = N >> 4;                       // B fragments per k-step
  const int numStrips = (M >> 4) * stripsN;
  const int strip = blockIdx.x * (blockDim.x / LANES) + wave;
  if (strip >= numStrips) return;                   // whole-wave exit: EXEC stays all-ones

  const int tm = strip / stripsN;
  const int ns = strip - tm * stripsN;
  const int r16  = lane & 15;                       // A: row-in-tile; C: col-in-subtile
  const int half = lane >> 4;                       // K half (0: K 0..15, 1: K 16..31)

  const int rowA = tm * 16 + r16;
  const float* arow  = A + (size_t)rowA * K;
  const float* a2row = HAS_A2 ? (A2 + (size_t)rowA * K) : nullptr;
  const int colBase = ns * 64 + r16;

  v8f zero = {};
  v8f acc[4];
#pragma unroll
  for (int t = 0; t < 4; ++t) acc[t] = zero;

#pragma unroll
  for (int ks = 0; ks < K; ks += 32) {
    const int kbase = ks + half * 16;

    // A tile (16x32 f16): lane r16(+16*half) holds row rowA, K kbase..kbase+15
    v16h a;
    const float4* ap = reinterpret_cast<const float4*>(arow + kbase);
#pragma unroll
    for (int i = 0; i < 4; ++i) {
      float4 f = ap[i];
      if constexpr (HAS_A2) {
        float4 g = reinterpret_cast<const float4*>(a2row + kbase)[i];
        f.x += g.x; f.y += g.y; f.z += g.z; f.w += g.w;
      }
      a[4 * i + 0] = (_Float16)f.x;
      a[4 * i + 1] = (_Float16)f.y;
      a[4 * i + 2] = (_Float16)f.z;
      a[4 * i + 3] = (_Float16)f.w;
    }

    // 4 pre-packed B subtiles reuse the A fragment (one 32B load each)
#pragma unroll
    for (int t = 0; t < 4; ++t) {
      const v16h b = Bp[((size_t)(ks >> 5) * fragsN + (ns * 4 + t)) * 32 + lane];
      acc[t] = __builtin_amdgcn_wmma_f32_16x16x32_f16(
          /*neg_a=*/false, a, /*neg_b=*/false, b,
          /*c_mod=*/(short)0, acc[t], /*reuse_a=*/false, /*reuse_b=*/false);
    }
  }

  // C/D layout: VGPR v -> M = v + 8*half ; N = r16 (within each 16-col subtile)
#pragma unroll
  for (int t = 0; t < 4; ++t) {
#pragma unroll
    for (int v = 0; v < 8; ++v) {
      const int r = tm * 16 + v + 8 * half;
      C[(size_t)r * N + colBase + t * 16] = acc[t][v];
    }
  }
}

// ---------------- edge scatter: out[col,:] += norm[e] * xw[row,:] ----------------
// block = F threads handles one edge; one f32 atomic per (edge, feature).
__global__ void spmm_scatter_kernel(const int* __restrict__ row, const int* __restrict__ col,
                                    const float* __restrict__ norm,
                                    const float* __restrict__ xw, float* __restrict__ out,
                                    int F) {
  const int e = blockIdx.x;
  const int f = threadIdx.x;
  const float nm = norm[e];
  if (nm == 0.0f) return;
  const int r = row[e];
  const int c = col[e];
  atomicAdd(&out[(size_t)c * F + f], nm * xw[(size_t)r * F + f]);
}

// ---------------- finalize: out = act(out + dinv^2 * xw + bias) ----------------
// block = F threads, grid = N nodes. act: 0 = relu, 1 = sigmoid.
__global__ void finalize_kernel(float* __restrict__ out, const float* __restrict__ xw,
                                const float* __restrict__ dinv, const float* __restrict__ bias,
                                int F, int act) {
  const int node = blockIdx.x;
  const int f = threadIdx.x;
  const size_t idx = (size_t)node * F + f;
  const float di = dinv[node];
  float v = out[idx] + di * di * xw[idx] + bias[f];
  if (act == 0) {
    v = v > 0.0f ? v : 0.0f;
  } else {
    v = 1.0f / (1.0f + expf(-v));
  }
  out[idx] = v;
}

// ---------------- edge_probs[e] = nan_to_num(sigmoid(dot(z[row], z[col]))) -----
// wave (32 lanes) per edge; F = 128 -> 4 f32 per lane, shfl_xor reduce.
__global__ void edge_dot_kernel(const float* __restrict__ z, const int* __restrict__ row,
                                const int* __restrict__ col, float* __restrict__ probs,
                                int E) {
  const int wavesPerBlock = blockDim.x / LANES;
  const int e = blockIdx.x * wavesPerBlock + (threadIdx.x / LANES);
  if (e >= E) return;
  const int lane = threadIdx.x & (LANES - 1);
  const float* zr = z + (size_t)row[e] * 128;
  const float* zc = z + (size_t)col[e] * 128;
  float s = 0.0f;
#pragma unroll
  for (int j = 0; j < 4; ++j) {
    const int idx = lane + 32 * j;
    s += zr[idx] * zc[idx];
  }
#pragma unroll
  for (int off = 16; off > 0; off >>= 1) s += __shfl_xor(s, off, LANES);
  if (lane == 0) {
    float p = 1.0f / (1.0f + expf(-s));
    if (isnan(p)) p = 0.0f;
    probs[e] = p;
  }
}

// ---------------- fc: accum += dot(d, wfc) (wave-reduced atomics) ----------------
__global__ void fc_dot_kernel(const float* __restrict__ d, const float* __restrict__ w,
                              long long n, float* __restrict__ accum) {
  long long i = (long long)blockIdx.x * blockDim.x + threadIdx.x;
  long long stride = (long long)gridDim.x * blockDim.x;
  float s = 0.0f;
  for (; i < n; i += stride) s += d[i] * w[i];
#pragma unroll
  for (int off = 16; off > 0; off >>= 1) s += __shfl_xor(s, off, LANES);
  if ((threadIdx.x & (LANES - 1)) == 0) atomicAdd(accum, s);
}

__global__ void final_sigmoid_kernel(const float* __restrict__ accum,
                                     const float* __restrict__ bfc,
                                     float* __restrict__ out) {
  out[0] = 1.0f / (1.0f + expf(-(accum[0] + bfc[0])));
}

// ---------------------------------------------------------------------------
extern "C" void kernel_launch(void* const* d_in, const int* in_sizes, int n_in,
                              void* d_out, int out_size, void* d_ws, size_t ws_size,
                              hipStream_t stream) {
  (void)n_in; (void)out_size; (void)ws_size;

  const float* features = (const float*)d_in[0];
  const int*   eidx     = (const int*)d_in[1];   // [2, E] flattened; harness int32
  const float* eattr    = (const float*)d_in[2];
  const float* Wg1      = (const float*)d_in[3];
  const float* bg1      = (const float*)d_in[4];
  const float* Wg2      = (const float*)d_in[5];
  const float* bg2      = (const float*)d_in[6];
  const float* Wd1      = (const float*)d_in[7];
  const float* bd1      = (const float*)d_in[8];
  const float* wfc      = (const float*)d_in[9];
  const float* bfc      = (const float*)d_in[10];

  const int FIN = 128, HID = 64;
  const int N = in_sizes[0] / FIN;     // 50000
  const int E = in_sizes[2];           // 1600000
  const int* row = eidx;
  const int* col = eidx + E;

  // ---- workspace layout (floats; all offsets 64B-aligned) ----
  float* ws = (float*)d_ws;
  size_t off = 0;
  float* xw1   = ws + off; off += (size_t)N * HID;   // features @ Wg1
  float* out1  = ws + off; off += (size_t)N * HID;   // h
  float* xw2   = ws + off; off += (size_t)N * FIN;   // h @ Wg2
  float* out2  = ws + off; off += (size_t)N * FIN;   // z
  float* xw3   = ws + off; off += (size_t)N * HID;   // (z+feat) @ Wd1
  float* out3  = ws + off; off += (size_t)N * HID;   // d
  float* norm1 = ws + off; off += (size_t)E;
  float* probs = ws + off; off += (size_t)E;
  float* norm3 = ws + off; off += (size_t)E;
  float* deg1  = ws + off; off += (size_t)N;         // becomes dinv1
  float* deg3  = ws + off; off += (size_t)N;         // becomes dinv3
  float* accum = ws + off; off += 16;
  _Float16* Bp1 = (_Float16*)(ws + off); off += (size_t)FIN * HID / 2;  // packed Wg1
  _Float16* Bp2 = (_Float16*)(ws + off); off += (size_t)HID * FIN / 2;  // packed Wg2
  _Float16* Bp3 = (_Float16*)(ws + off); off += (size_t)FIN * HID / 2;  // packed Wd1

  const int TB = 256;
  const int GS = 2048;  // grid-stride grid

  // ---- init scratch (harness poisons ws with 0xAA) ----
  fill_kernel<<<GS, TB, 0, stream>>>(out1, (long long)N * HID, 0.0f);
  fill_kernel<<<GS, TB, 0, stream>>>(out2, (long long)N * FIN, 0.0f);
  fill_kernel<<<GS, TB, 0, stream>>>(out3, (long long)N * HID, 0.0f);
  fill_kernel<<<GS, TB, 0, stream>>>(deg1, (long long)2 * N, 1.0f); // deg1+deg3, self-loop=1
  fill_kernel<<<1, 32, 0, stream>>>(accum, 16, 0.0f);

  // ---- pack weights into WMMA B-fragment layout (f16) ----
  pack_b_kernel<<<2, TB, 0, stream>>>(Wg1, Bp1, HID, FIN);  // 512 frag-lanes
  pack_b_kernel<<<2, TB, 0, stream>>>(Wg2, Bp2, FIN, HID);
  pack_b_kernel<<<2, TB, 0, stream>>>(Wd1, Bp3, HID, FIN);

  // ---- conv1/conv2 shared normalization (same edge weights) ----
  deg_kernel<<<GS, TB, 0, stream>>>(col, eattr, deg1, E);
  dinv_kernel<<<GS, TB, 0, stream>>>(deg1, N);
  norm_kernel<<<GS, TB, 0, stream>>>(row, col, eattr, deg1, norm1, E);

  const int wavesPerBlock = TB / LANES;   // 8 waves -> 8 strips per block

  // ---- generator conv1: h = relu(prop(features @ Wg1) + bg1) ----
  {
    int strips = (N / 16) * (HID / 64);
    int blocks = (strips + wavesPerBlock - 1) / wavesPerBlock;
    gemm_wmma_kernel<128, false><<<blocks, TB, 0, stream>>>(
        features, nullptr, (const v16h*)Bp1, xw1, N, HID);
  }
  spmm_scatter_kernel<<<E, HID, 0, stream>>>(row, col, norm1, xw1, out1, HID);
  finalize_kernel<<<N, HID, 0, stream>>>(out1, xw1, deg1, bg1, HID, /*relu*/0);

  // ---- generator conv2: z = sigmoid(prop(h @ Wg2) + bg2) ----
  {
    int strips = (N / 16) * (FIN / 64);
    int blocks = (strips + wavesPerBlock - 1) / wavesPerBlock;
    gemm_wmma_kernel<64, false><<<blocks, TB, 0, stream>>>(
        out1, nullptr, (const v16h*)Bp2, xw2, N, FIN);
  }
  spmm_scatter_kernel<<<E, FIN, 0, stream>>>(row, col, norm1, xw2, out2, FIN);
  finalize_kernel<<<N, FIN, 0, stream>>>(out2, xw2, deg1, bg2, FIN, /*sigmoid*/1);

  // ---- decoder: edge_probs = nan_to_num(sigmoid(<z[row], z[col]>)) ----
  {
    int blocks = (E + wavesPerBlock - 1) / wavesPerBlock;
    edge_dot_kernel<<<blocks, TB, 0, stream>>>(out2, row, col, probs, E);
  }

  // ---- discriminator normalization with edge_probs as weights ----
  deg_kernel<<<GS, TB, 0, stream>>>(col, probs, deg3, E);
  dinv_kernel<<<GS, TB, 0, stream>>>(deg3, N);
  norm_kernel<<<GS, TB, 0, stream>>>(row, col, probs, deg3, norm3, E);

  // ---- discriminator conv: d = relu(prop((z+features) @ Wd1) + bd1) ----
  // enc = z + features folded into the GEMM A-operand load.
  {
    int strips = (N / 16) * (HID / 64);
    int blocks = (strips + wavesPerBlock - 1) / wavesPerBlock;
    gemm_wmma_kernel<128, true><<<blocks, TB, 0, stream>>>(
        out2, features, (const v16h*)Bp3, xw3, N, HID);
  }
  spmm_scatter_kernel<<<E, HID, 0, stream>>>(row, col, norm3, xw3, out3, HID);
  finalize_kernel<<<N, HID, 0, stream>>>(out3, xw3, deg3, bd1, HID, /*relu*/0);

  // ---- final FC + sigmoid ----
  fc_dot_kernel<<<GS, TB, 0, stream>>>(out3, wfc, (long long)N * HID, accum);
  final_sigmoid_kernel<<<1, 1, 0, stream>>>(accum, bfc, (float*)d_out);
}
